// RotaryMultiHeadAttention_11063835755227
// MI455X (gfx1250) — matrix operand order repaired and checked by reading, not compiled, not run on hardware
//
#include <hip/hip_runtime.h>

// ---------------------------------------------------------------------------
// Rotary Multi-Head Attention for MI455X (gfx1250, wave32, WMMA f16->f32)
// B=2, L=2048, E=1024, H=16, HD=64
// ---------------------------------------------------------------------------

typedef __attribute__((ext_vector_type(16))) _Float16 v16h;
typedef __attribute__((ext_vector_type(8)))  _Float16 v8h;
typedef __attribute__((ext_vector_type(8)))  float    v8f;

constexpr int B_  = 2;
constexpr int L_  = 2048;
constexpr int E_  = 1024;
constexpr int H_  = 16;
constexpr int HD_ = 64;
constexpr float ATTN_SCALE = 0.125f; // 1/sqrt(64), folded into Q at RoPE time

__device__ __forceinline__ v8f wmma_f16(v16h a, v16h b, v8f c) {
  // v_wmma_f32_16x16x32_f16: D = A(16x32) x B(32x16) + C(16x16 f32)
  return __builtin_amdgcn_wmma_f32_16x16x32_f16(
      /*neg_a=*/false, a, /*neg_b=*/false, b,
      /*c_mod=*/(short)0, c, /*reuse_a=*/false, /*reuse_b=*/false);
}

// A fragment (16x32, 16-bit): lane = {M = lane&15, half = lane>>4};
// elements 0..7 -> K = 8*half + e, elements 8..15 -> K = 16 + 8*half + e.
// base points at (row0, k0) of a row-major [M x K] f16 matrix, ld in halfs.
__device__ __forceinline__ v16h load_a_frag(const _Float16* base, int ld, int lane) {
  const int m  = lane & 15;
  const int hf = lane >> 4;
  const _Float16* r = base + m * ld + 8 * hf;
  v8h lo = *(const v8h*)(r);        // K = 8*hf + 0..7
  v8h hi = *(const v8h*)(r + 16);   // K = 16 + 8*hf + 0..7
  v16h a;
#pragma unroll
  for (int e = 0; e < 8; ++e) { a[e] = lo[e]; a[e + 8] = hi[e]; }
  return a;
}

// B fragment (32x16, 16-bit): lane = {N = lane&15, half = lane>>4};
// elements 0..7 -> K = 8*half + e, elements 8..15 -> K = 16 + 8*half + e.
// base points at (k0, n0) of a row-major [K x N] f16 matrix, ld in halfs.
__device__ __forceinline__ v16h load_b_frag(const _Float16* base, int ld, int lane) {
  const int n  = lane & 15;
  const int hf = lane >> 4;
  v16h b;
#pragma unroll
  for (int e = 0; e < 8; ++e) {
    b[e]     = base[(8 * hf + e) * ld + n];
    b[e + 8] = base[(16 + 8 * hf + e) * ld + n];
  }
  return b;
}

__device__ __forceinline__ float rowmax16(float x) {
  x = fmaxf(x, __shfl_xor(x, 1, 32));
  x = fmaxf(x, __shfl_xor(x, 2, 32));
  x = fmaxf(x, __shfl_xor(x, 4, 32));
  x = fmaxf(x, __shfl_xor(x, 8, 32));
  return x;
}
__device__ __forceinline__ float rowsum16(float x) {
  x += __shfl_xor(x, 1, 32);
  x += __shfl_xor(x, 2, 32);
  x += __shfl_xor(x, 4, 32);
  x += __shfl_xor(x, 8, 32);
  return x;
}

// ---------------------------------------------------------------------------
// Elementwise prep kernels
// ---------------------------------------------------------------------------
__global__ void rmha_cvt_f16(const float* __restrict__ src,
                             _Float16* __restrict__ dst, int n) {
  int i = blockIdx.x * blockDim.x + threadIdx.x;
  if (i < n) dst[i] = (_Float16)src[i];
}

// Wt[k][n] = W[n][k], f32 -> f16. W is [E x E] row-major.
__global__ void rmha_transpose_w(const float* __restrict__ W,
                                 _Float16* __restrict__ Wt) {
  int i = blockIdx.x * blockDim.x + threadIdx.x; // i = n*E + k
  int k = i & (E_ - 1);
  int n = i >> 10;
  Wt[k * E_ + n] = (_Float16)W[n * E_ + k];
}

// ---------------------------------------------------------------------------
// GEMM: C[M,1024] = A_h[M,1024] @ Bt_h[1024,1024] + bias
// Wave tile 32x64: 2 M-frags x 4 N-frags = 8 WMMA accumulators.
// Ping-pong pipelined (unroll-by-2, two named fragment sets): each half
// iteration loads the *other* buffer, so there is no register rotation
// (no v_dual_mov chains, no WMMA->VALU WAR hazard NOPs) and loads retire
// a full K-step ahead of their consuming WMMAs. __launch_bounds__(256,1)
// keeps the ~180-VGPR working set in registers (no scratch spills).
// ---------------------------------------------------------------------------
struct Frags {
  v16h a0, a1;       // two 16x32 A tiles (rows row0, row0+16)
  v16h b0, b1, b2, b3; // four 32x16 B tiles (cols col0 + 0/16/32/48)
};

__device__ __forceinline__ Frags load_frags(const _Float16* __restrict__ Abase,
                                            const _Float16* __restrict__ Bbase,
                                            int k0, int lane) {
  constexpr int K = E_;
  constexpr int N = E_;
  Frags f;
  f.a0 = load_a_frag(Abase + k0, K, lane);
  f.a1 = load_a_frag(Abase + (size_t)16 * K + k0, K, lane);
  f.b0 = load_b_frag(Bbase + (size_t)k0 * N,      N, lane);
  f.b1 = load_b_frag(Bbase + (size_t)k0 * N + 16, N, lane);
  f.b2 = load_b_frag(Bbase + (size_t)k0 * N + 32, N, lane);
  f.b3 = load_b_frag(Bbase + (size_t)k0 * N + 48, N, lane);
  return f;
}

__device__ __forceinline__ void mm_step(const Frags& f, v8f acc[2][4]) {
  acc[0][0] = wmma_f16(f.a0, f.b0, acc[0][0]);
  acc[1][0] = wmma_f16(f.a1, f.b0, acc[1][0]);
  acc[0][1] = wmma_f16(f.a0, f.b1, acc[0][1]);
  acc[1][1] = wmma_f16(f.a1, f.b1, acc[1][1]);
  acc[0][2] = wmma_f16(f.a0, f.b2, acc[0][2]);
  acc[1][2] = wmma_f16(f.a1, f.b2, acc[1][2]);
  acc[0][3] = wmma_f16(f.a0, f.b3, acc[0][3]);
  acc[1][3] = wmma_f16(f.a1, f.b3, acc[1][3]);
}

__global__ void __launch_bounds__(256, 1)
rmha_gemm(const _Float16* __restrict__ A,
          const _Float16* __restrict__ Bt,
          const float* __restrict__ bias,
          float* __restrict__ Cf,        // f32 out (or null)
          _Float16* __restrict__ Ch) {   // f16 out (or null)
  constexpr int K = E_;
  constexpr int N = E_;
  const int lane = threadIdx.x & 31;
  const int wave = threadIdx.x >> 5;
  const int wm = wave >> 1;          // 0..3
  const int wn = wave & 1;           // 0..1
  const int row0 = blockIdx.x * 128 + wm * 32;
  const int col0 = blockIdx.y * 128 + wn * 64;

  const _Float16* Abase = A + (size_t)row0 * K;
  const _Float16* Bbase = Bt + col0;

  v8f acc[2][4] = {};
  Frags f0 = load_frags(Abase, Bbase, 0, lane);

#pragma unroll 1
  for (int k0 = 0; k0 < K - 64; k0 += 64) {
    __builtin_prefetch(Abase + (size_t)(lane & 15) * K + k0 + 96, 0, 1);
    __builtin_prefetch(Bbase + (size_t)(k0 + 96 + lane) * N, 0, 1);

    Frags f1 = load_frags(Abase, Bbase, k0 + 32, lane); // fill other buffer
    mm_step(f0, acc);                                   // consume f0 (k0)
    f0 = load_frags(Abase, Bbase, k0 + 64, lane);       // refill f0
    mm_step(f1, acc);                                   // consume f1 (k0+32)
  }
  // Tail: two remaining K-steps (K-64 in f0, K-32 loaded here)
  {
    Frags f1 = load_frags(Abase, Bbase, K - 32, lane);
    mm_step(f0, acc);
    mm_step(f1, acc);
  }

  // C layout: lane -> {N = lane&15, mhalf = lane>>4}; VGPR j -> M = j + 8*mhalf
  const int n  = lane & 15;
  const int mh = lane >> 4;
#pragma unroll
  for (int i = 0; i < 2; ++i) {
#pragma unroll
    for (int jn = 0; jn < 4; ++jn) {
      const int c = col0 + jn * 16 + n;
      const float bv = bias[c];
#pragma unroll
      for (int j = 0; j < 8; ++j) {
        const int r = row0 + i * 16 + j + 8 * mh;
        const float val = acc[i][jn][j] + bv;
        if (Cf) Cf[(size_t)r * N + c] = val;
        else    Ch[(size_t)r * N + c] = (_Float16)val;
      }
    }
  }
}

// ---------------------------------------------------------------------------
// RoPE (interleaved pairs) + head-major repack.
// Input Y_h [B*L, E] (row = b*L + l, col = h*HD + d), f16 (bias included).
// out_rows: [bh][l][d] (Q path, scaled by 1/sqrt(HD));  out_cols: [bh][d][l]
// (K path, transposed so S=Q.K^T B-fragments are contiguous along l).
// ---------------------------------------------------------------------------
__global__ void rmha_rope_pack(const _Float16* __restrict__ Y,
                               _Float16* __restrict__ out_rows,
                               _Float16* __restrict__ out_cols,
                               float scale) {
  const int tid = blockIdx.x * blockDim.x + threadIdx.x; // (bh, l, pair)
  const int i  = tid & 31;            // pair index, d = 2i, 2i+1
  const int l  = (tid >> 5) & (L_ - 1);
  const int bh = tid >> 16;
  const int b = bh >> 4, h = bh & (H_ - 1);
  const int d = 2 * i;

  const size_t src = ((size_t)b * L_ + l) * E_ + h * HD_ + d;
  const float x0 = (float)Y[src];
  const float x1 = (float)Y[src + 1];

  const float inv_freq = __powf(10000.0f, -(float)d / (float)HD_);
  const float f = (float)l * inv_freq;
  const float c = __cosf(f), s = __sinf(f);

  const float o0 = (x0 * c - x1 * s) * scale;
  const float o1 = (x1 * c + x0 * s) * scale;

  if (out_rows) {
    const size_t dst = ((size_t)bh * L_ + l) * HD_ + d;
    out_rows[dst]     = (_Float16)o0;
    out_rows[dst + 1] = (_Float16)o1;
  }
  if (out_cols) {
    out_cols[((size_t)bh * HD_ + d) * L_ + l]     = (_Float16)o0;
    out_cols[((size_t)bh * HD_ + d + 1) * L_ + l] = (_Float16)o1;
  }
}

// V repack: [B*L, E] -> [bh][l][d]
__global__ void rmha_v_pack(const _Float16* __restrict__ Y,
                            _Float16* __restrict__ vh) {
  const int tid = blockIdx.x * blockDim.x + threadIdx.x; // (bh, l, d)
  const int d  = tid & (HD_ - 1);
  const int l  = (tid >> 6) & (L_ - 1);
  const int bh = tid >> 17;
  const int b = bh >> 4, h = bh & (H_ - 1);
  vh[tid] = Y[((size_t)b * L_ + l) * E_ + h * HD_ + d];
}

// ---------------------------------------------------------------------------
// Flash attention: one wave owns one (bh, 16-row Q tile); streams K-blocks
// of 32 columns. S via WMMA, online softmax with wave32 shuffles, P staged
// C-layout -> A-layout through per-wave LDS (s_wait_dscnt), P.V via WMMA.
// V fragments are loaded before the softmax so their latency hides behind
// the shuffle/exp VALU work. Output f16 to aout [B*L, E].
// ---------------------------------------------------------------------------
__global__ void __launch_bounds__(256, 1)
rmha_attention(const _Float16* __restrict__ qh,  // [bh][l][d]
               const _Float16* __restrict__ kt,  // [bh][d][l]
               const _Float16* __restrict__ vh,  // [bh][l][d]
               _Float16* __restrict__ aout) {    // [B*L, E]
  __shared__ _Float16 Plds[8][16][32]; // per-wave P staging (8 KB)

  const int lane = threadIdx.x & 31;
  const int wave = threadIdx.x >> 5;
  const int tile = blockIdx.x * 8 + wave;      // B*H*(L/16) = 4096 tiles
  const int bh = tile >> 7;                    // / (L/16 = 128)
  const int qt = tile & 127;
  const int qrow0 = qt * 16;
  const int b = bh >> 4, h = bh & (H_ - 1);

  const _Float16* qb = qh + (size_t)bh * L_ * HD_;
  const _Float16* kb = kt + (size_t)bh * HD_ * L_;
  const _Float16* vb = vh + (size_t)bh * L_ * HD_;

  // Q fragments are loop-invariant: A tiles for K-dim slabs d=[0,32),[32,64)
  v16h aq[2];
  aq[0] = load_a_frag(qb + (size_t)qrow0 * HD_, HD_, lane);
  aq[1] = load_a_frag(qb + (size_t)qrow0 * HD_ + 32, HD_, lane);

  v8f O[4] = {};
  float m[8], lsum[8];
#pragma unroll
  for (int j = 0; j < 8; ++j) { m[j] = -INFINITY; lsum[j] = 0.0f; }

  const int nlan = lane & 15;
  const int hf   = lane >> 4;
  _Float16* pw = &Plds[wave][0][0];

  for (int kblk = 0; kblk < L_; kblk += 32) {
    // ---- S = (scaled Q) . K^T for 32 key positions -> two 16x16 f32 tiles
    v8f s0 = {}, s1 = {};
#pragma unroll
    for (int ds = 0; ds < 2; ++ds) {
      v16h kb0 = load_b_frag(kb + (size_t)(ds * 32) * L_ + kblk,      L_, lane);
      v16h kb1 = load_b_frag(kb + (size_t)(ds * 32) * L_ + kblk + 16, L_, lane);
      s0 = wmma_f16(aq[ds], kb0, s0);
      s1 = wmma_f16(aq[ds], kb1, s1);
    }

    // ---- V B-fragments for this key block: issue now, consume after
    // softmax so the loads overlap the VALU-heavy online-softmax update.
    v16h vf[4];
#pragma unroll
    for (int nt = 0; nt < 4; ++nt)
      vf[nt] = load_b_frag(vb + (size_t)kblk * HD_ + nt * 16, HD_, lane);

    // ---- online softmax; C layout: VGPR j holds row (j + 8*hf), N = nlan
#pragma unroll
    for (int j = 0; j < 8; ++j) {
      const float t    = fmaxf(s0[j], s1[j]);
      const float mloc = rowmax16(t);
      const float mnew = fmaxf(m[j], mloc);
      const float alpha = __expf(m[j] - mnew);
      const float p0 = __expf(s0[j] - mnew);
      const float p1 = __expf(s1[j] - mnew);
      lsum[j] = lsum[j] * alpha + rowsum16(p0 + p1);
      m[j] = mnew;
#pragma unroll
      for (int nt = 0; nt < 4; ++nt) O[nt][j] *= alpha;
      const int row = j + 8 * hf;                   // P row (query) in tile
      pw[row * 32 + nlan]      = (_Float16)p0;      // key col = nlan
      pw[row * 32 + 16 + nlan] = (_Float16)p1;      // key col = 16 + nlan
    }

    // CDNA5 split dependency counter: ensure the DS stores landed before
    // re-reading P in A-fragment order.
    asm volatile("s_wait_dscnt 0" ::: "memory");

    // ---- P (16x32 f16, A layout) from LDS
    {
      const int ma = lane & 15;
      v8h lo = *(const v8h*)(pw + ma * 32 + 8 * hf);
      v8h hi = *(const v8h*)(pw + ma * 32 + 16 + 8 * hf);
      v16h pa;
#pragma unroll
      for (int e = 0; e < 8; ++e) { pa[e] = lo[e]; pa[e + 8] = hi[e]; }

      // ---- O += P . V   (V B-fragments: lane = head dim, elems = key pos)
#pragma unroll
      for (int nt = 0; nt < 4; ++nt)
        O[nt] = wmma_f16(pa, vf[nt], O[nt]);
    }
  }

  // ---- normalize and scatter to [B*L, E] f16 for the output projection
#pragma unroll
  for (int j = 0; j < 8; ++j) {
    const float inv = 1.0f / lsum[j];
    const int row = qrow0 + j + 8 * hf;
    const size_t orow = ((size_t)b * L_ + row) * E_ + h * HD_;
#pragma unroll
    for (int nt = 0; nt < 4; ++nt)
      aout[orow + nt * 16 + nlan] = (_Float16)(O[nt][j] * inv);
  }
}

// ---------------------------------------------------------------------------
// Host-side orchestration
// ---------------------------------------------------------------------------
extern "C" void kernel_launch(void* const* d_in, const int* in_sizes, int n_in,
                              void* d_out, int out_size, void* d_ws, size_t ws_size,
                              hipStream_t stream) {
  (void)in_sizes; (void)n_in; (void)out_size; (void)ws_size;
  const float* x  = (const float*)d_in[0];
  const float* Wq = (const float*)d_in[1];
  const float* bq = (const float*)d_in[2];
  const float* Wk = (const float*)d_in[3];
  const float* bk = (const float*)d_in[4];
  const float* Wv = (const float*)d_in[5];
  const float* bv = (const float*)d_in[6];
  const float* Wo = (const float*)d_in[7];
  const float* bo = (const float*)d_in[8];
  float* out = (float*)d_out;

  const int M = B_ * L_;               // 4096
  char* w = (char*)d_ws;
  auto take = [&](size_t bytes) {
    void* p = (void*)w;
    w += (bytes + 255) & ~(size_t)255;
    return p;
  };
  _Float16* xh   = (_Float16*)take((size_t)M * E_ * 2);        // x as f16
  _Float16* WtQ  = (_Float16*)take((size_t)E_ * E_ * 2);
  _Float16* WtK  = (_Float16*)take((size_t)E_ * E_ * 2);
  _Float16* WtV  = (_Float16*)take((size_t)E_ * E_ * 2);
  _Float16* WtO  = (_Float16*)take((size_t)E_ * E_ * 2);
  _Float16* Yq   = (_Float16*)take((size_t)M * E_ * 2);        // x@Wq^T + bq
  _Float16* Yk   = (_Float16*)take((size_t)M * E_ * 2);
  _Float16* Yv   = (_Float16*)take((size_t)M * E_ * 2);
  _Float16* qhh  = (_Float16*)take((size_t)M * E_ * 2);        // roped Q [bh][l][d]
  _Float16* ktt  = (_Float16*)take((size_t)M * E_ * 2);        // roped K [bh][d][l]
  _Float16* vhh  = (_Float16*)take((size_t)M * E_ * 2);        // V [bh][l][d]
  _Float16* aout = (_Float16*)take((size_t)M * E_ * 2);        // attn out [B*L, E]

  const dim3 blk(256);

  // 1) x -> f16; weights -> transposed f16
  rmha_cvt_f16<<<(M * E_) / 256, blk, 0, stream>>>(x, xh, M * E_);
  rmha_transpose_w<<<(E_ * E_) / 256, blk, 0, stream>>>(Wq, WtQ);
  rmha_transpose_w<<<(E_ * E_) / 256, blk, 0, stream>>>(Wk, WtK);
  rmha_transpose_w<<<(E_ * E_) / 256, blk, 0, stream>>>(Wv, WtV);
  rmha_transpose_w<<<(E_ * E_) / 256, blk, 0, stream>>>(Wo, WtO);

  // 2) QKV projections (WMMA), f16 out with bias; 128x128 block tiles
  const dim3 ggrid(M / 128, E_ / 128);
  rmha_gemm<<<ggrid, blk, 0, stream>>>(xh, WtQ, bq, nullptr, Yq);
  rmha_gemm<<<ggrid, blk, 0, stream>>>(xh, WtK, bk, nullptr, Yk);
  rmha_gemm<<<ggrid, blk, 0, stream>>>(xh, WtV, bv, nullptr, Yv);

  // 3) RoPE + head-major repack (Q scaled by 1/sqrt(HD); K transposed)
  const int rope_threads = B_ * H_ * L_ * (HD_ / 2);           // 2,097,152
  rmha_rope_pack<<<rope_threads / 256, blk, 0, stream>>>(Yq, qhh, nullptr, ATTN_SCALE);
  rmha_rope_pack<<<rope_threads / 256, blk, 0, stream>>>(Yk, nullptr, ktt, 1.0f);
  rmha_v_pack<<<(M * E_) / 256, blk, 0, stream>>>(Yv, vhh);

  // 4) flash attention: 4096 wave-tiles / 8 waves per block
  rmha_attention<<<(B_ * H_ * (L_ / 16)) / 8, blk, 0, stream>>>(qhh, ktt, vhh, aout);

  // 5) output projection, f32 + bias -> d_out
  rmha_gemm<<<ggrid, blk, 0, stream>>>(aout, WtO, bo, out, nullptr);
}
